// RelativeMultiheadAttention_69312182223123
// MI455X (gfx1250) — matrix operand order repaired and checked by reading, not compiled, not run on hardware
//
#include <hip/hip_runtime.h>
#include <hip/hip_bf16.h>
#include <stdint.h>

// ---------------------------------------------------------------------------
// Relative multi-head attention (sliding window CTX=64), MI455X / gfx1250.
// v3: all fp32->bf16 conversion hoisted into one-time prep kernels; GEMM and
// attention staging use GLOBAL_LOAD_ASYNC_TO_LDS_B128 (pure bf16 copies),
// WMMA core = v_wmma_f32_16x16x32_bf16 with fp32 accumulation.
// Workspace: ~24 MB of bf16 mirrors (assumed <= ws_size).
// ---------------------------------------------------------------------------

#define N_HEADC 8
#define D_HEADC 64
#define D_MODELC 512
#define CTXC 64
#define BATCHC 2
#define Q_LENC 2048
#define T_LENC (Q_LENC + CTXC - 1) /* 2111 */
#define HDC (N_HEADC * D_HEADC)    /* 512 */

typedef __attribute__((ext_vector_type(16))) __bf16 v16bf;
typedef __attribute__((ext_vector_type(8)))  float  v8f;
typedef int v4i __attribute__((vector_size(16))); // matches async-lds builtin

union BF16Frag {
  v16bf v;
  unsigned short s[16];
  uint4 q[2];
};

__device__ __forceinline__ unsigned short f2bf(float f) {
  union { float f; unsigned u; } x; x.f = f;
  unsigned r = x.u + 0x7FFFu + ((x.u >> 16) & 1u); // round-to-nearest-even
  return (unsigned short)(r >> 16);
}
__device__ __forceinline__ unsigned packbf2(float lo, float hi) {
  return (unsigned)f2bf(lo) | ((unsigned)f2bf(hi) << 16);
}

__device__ __forceinline__ v8f wmma_bf16(const BF16Frag& a, const BF16Frag& b, v8f c) {
  return __builtin_amdgcn_wmma_f32_16x16x32_bf16(false, a.v, false, b.v,
                                                 (short)0, c, false, false);
}

// ---- async global->LDS copy (CDNA5), 16 bytes per lane -------------------
#ifdef __has_builtin
#if __has_builtin(__builtin_amdgcn_global_load_async_to_lds_b128)
#define HAVE_ASYNC_LDS 1
#endif
#endif
#ifndef HAVE_ASYNC_LDS
#define HAVE_ASYNC_LDS 0
#endif

__device__ __forceinline__ void async_cp16(const void* g, void* l) {
#if HAVE_ASYNC_LDS
  __builtin_amdgcn_global_load_async_to_lds_b128((v4i*)g, (v4i*)l, 0, 0);
#else
  *(uint4*)l = *(const uint4*)g;
#endif
}
__device__ __forceinline__ void async_wait() {
#if HAVE_ASYNC_LDS
  asm volatile("s_wait_asynccnt 0x0" ::: "memory");
#endif
}

// ---------------------------------------------------------------------------
// Prep kernels: fp32 -> bf16 conversion / weight transpose, pad mask.
// ---------------------------------------------------------------------------
__global__ void cvt_bf16_kernel(const float* __restrict__ src,
                                unsigned short* __restrict__ dst, int n4) {
  int i = blockIdx.x * blockDim.x + threadIdx.x; // group of 4 elements
  if (i >= n4) return;
  float4 f = *(const float4*)(src + (size_t)i * 4);
  uint2 u;
  u.x = packbf2(f.x, f.y);
  u.y = packbf2(f.z, f.w);
  *(uint2*)(dst + (size_t)i * 4) = u;
}

// WT[n][k] = bf16(W[k][n]); K, N multiples of 32. 256 threads, 32x32 tiles.
__global__ void transpose_bf16_kernel(const float* __restrict__ W,
                                      unsigned short* __restrict__ WT,
                                      int K, int N) {
  __shared__ float tile[32][33];
  const int n0 = blockIdx.x * 32, k0 = blockIdx.y * 32;
  const int tx = threadIdx.x & 31, ty = threadIdx.x >> 5; // ty 0..7
#pragma unroll
  for (int s = 0; s < 32; s += 8)
    tile[ty + s][tx] = W[(size_t)(k0 + ty + s) * N + n0 + tx];
  __syncthreads();
#pragma unroll
  for (int s = 0; s < 32; s += 8)
    WT[(size_t)(n0 + ty + s) * K + k0 + tx] = f2bf(tile[tx][ty + s]);
}

// mask[b*63 + t] = 1.0f iff all of x[b, t, :] == 0, for t < CTX-1
__global__ void mask_kernel(const float* __restrict__ x, float* __restrict__ mask) {
  const int idx = blockIdx.x * blockDim.x + threadIdx.x;
  if (idx >= BATCHC * (CTXC - 1)) return;
  const int b = idx / (CTXC - 1);
  const int t = idx % (CTXC - 1);
  const float* p = x + ((size_t)b * T_LENC + t) * D_MODELC;
  int allz = 1;
  for (int i = 0; i < D_MODELC; ++i)
    if (p[i] != 0.f) { allz = 0; break; }
  mask[idx] = (float)allz;
}

// ---------------------------------------------------------------------------
// GEMM: out[M x N] = remap(A)[M x K] * Bt^T (+ bias). A and Bt are bf16,
// Bt is pre-transposed [N x K]. Outputs: fp32 C and/or bf16 Cbf (nullable).
// Row remap: a_row(r) = (r / rowsPerBatch)*batchStride
//                      + ((r % rowsPerBatch) + rowOffset)*K
// Requires N % 128 == 0, K % 32 == 0. 256 thr = 8 waves, 64x128 block tile.
// ---------------------------------------------------------------------------
#define TBM 64
#define TBN 128
#define TBK 32

__global__ __launch_bounds__(256) void gemm_bf16(
    const unsigned short* __restrict__ A, const unsigned short* __restrict__ Bt,
    const float* __restrict__ bias, float* __restrict__ C,
    unsigned short* __restrict__ Cbf,
    int M, int N, int K, int rowsPerBatch, long long batchStride, int rowOffset)
{
  __shared__ unsigned short Alds[TBM][TBK + 16]; // [m][k], 96B row stride
  __shared__ unsigned short Blds[TBN][TBK + 16]; // [n][k]

  const int tid  = threadIdx.x;
  const int wv   = tid >> 5;
  const int lane = tid & 31;
  const int sel  = lane >> 4;
  const int l16  = lane & 15;
  const int bm = blockIdx.y * TBM, bn = blockIdx.x * TBN;
  const int wm = (wv >> 2) * 32, wn = (wv & 3) * 32;

  v8f acc[2][2] = {};

  for (int k0 = 0; k0 < K; k0 += TBK) {
    // ---- stage A tile 64x32 bf16: one 16B async copy per thread ----
    {
      const int row  = tid >> 2;
      const int cc   = (tid & 3) * 8;
      const int grow = bm + row;
      if (grow < M) {
        const unsigned short* ap =
            A + (size_t)(grow / rowsPerBatch) * (size_t)batchStride
              + (size_t)((grow % rowsPerBatch) + rowOffset) * K + (k0 + cc);
        async_cp16(ap, &Alds[row][cc]);
      } else {
        uint4 z = make_uint4(0u, 0u, 0u, 0u);
        *(uint4*)&Alds[row][cc] = z;
      }
    }
    // ---- stage Bt tile 128x32 bf16: two 16B async copies per thread ----
#pragma unroll
    for (int s = 0; s < 2; ++s) {
      const int lin  = tid + s * 256;
      const int rowN = lin >> 2;
      const int cc   = (lin & 3) * 8;
      async_cp16(Bt + (size_t)(bn + rowN) * K + (k0 + cc), &Blds[rowN][cc]);
    }
    async_wait();
    __syncthreads();

    BF16Frag af[2], bfr[2];
#pragma unroll
    for (int mi = 0; mi < 2; ++mi) {
      const unsigned short* p = &Alds[wm + 16 * mi + l16][0];
      af[mi].q[0] = *(const uint4*)(p + 8 * sel);
      af[mi].q[1] = *(const uint4*)(p + 16 + 8 * sel);
    }
#pragma unroll
    for (int ni = 0; ni < 2; ++ni) {
      const unsigned short* p = &Blds[wn + 16 * ni + l16][0];
      bfr[ni].q[0] = *(const uint4*)(p + 8 * sel);
      bfr[ni].q[1] = *(const uint4*)(p + 16 + 8 * sel);
    }
#pragma unroll
    for (int mi = 0; mi < 2; ++mi)
#pragma unroll
      for (int ni = 0; ni < 2; ++ni)
        acc[mi][ni] = wmma_bf16(af[mi], bfr[ni], acc[mi][ni]);
    __syncthreads();
  }

  // ---- epilogue: D layout lane->N(l16), vgpr v -> M = v + 8*sel ----
#pragma unroll
  for (int mi = 0; mi < 2; ++mi)
#pragma unroll
    for (int ni = 0; ni < 2; ++ni)
#pragma unroll
      for (int v = 0; v < 8; ++v) {
        const int row = bm + wm + 16 * mi + v + 8 * sel;
        const int col = bn + wn + 16 * ni + l16;
        if (row < M) {
          float val = acc[mi][ni][v];
          if (bias) val += bias[col];
          if (C)   C[(size_t)row * N + col] = val;
          if (Cbf) Cbf[(size_t)row * N + col] = f2bf(val);
        }
      }
}

// ---------------------------------------------------------------------------
// Attention core. Grid = B * H * (Q_LEN/64), 128 threads (4 waves), each wave
// owns a 16-query tile. K rows / hr are async-copied bf16 into LDS; V staged
// transposed; hq fragments are direct bf16 global loads.
//   S[i][c] = hq_i . k_{t0+c}  (dense 16x80; band c in [i, i+63])
//   S[i][i+j] += hq_i . hr_j ; softmax over j ; out = banded W x V
// ---------------------------------------------------------------------------
__global__ __launch_bounds__(128) void attn_kernel(
    const unsigned short* __restrict__ kvbf, // (B, T, 2*HD) bf16 [k | v]
    const unsigned short* __restrict__ hqbf, // (B, Q_LEN, HD) bf16
    const unsigned short* __restrict__ hrbf, // (CTX, HD) bf16
    const float* __restrict__ maskb,         // (B, CTX-1)
    unsigned short* __restrict__ obf)        // (B, Q_LEN, HD) bf16
{
  __shared__ unsigned short k_lds[128][72];   // [key c][dh]     (B for QK^T)
  __shared__ unsigned short v_lds[64][152];   // [dh][key c] (T) (B for WV)
  __shared__ unsigned short hr_lds[64][72];   // [j][dh]         (B for Q.hr^T)
  __shared__ float          s_lds[4][16][80]; // per-wave fp32 scores
  __shared__ unsigned short wd_lds[4][16][104]; // per-wave banded softmax bf16

  const int tid  = threadIdx.x;
  const int wv   = tid >> 5;
  const int lane = tid & 31;
  const int sel  = lane >> 4;
  const int l16  = lane & 15;

  const int blk   = blockIdx.x;
  const int qblk  = blk & 31;
  const int h     = (blk >> 5) & 7;
  const int b     = blk >> 8;
  const int qbase = qblk * 64;
  const int t0    = qbase;

  // ---- K rows [t0, t0+127]: 1024 16B chunks, async ----
#pragma unroll
  for (int s = 0; s < 8; ++s) {
    const int lin  = tid + s * 128;
    const int c    = lin >> 3;          // 0..127
    const int cc   = (lin & 7) * 8;     // 0..56
    const int trow = t0 + c;
    if (trow < T_LENC) {
      async_cp16(kvbf + ((size_t)b * T_LENC + trow) * (2 * HDC) + h * D_HEADC + cc,
                 &k_lds[c][cc]);
    } else {
      uint4 z = make_uint4(0u, 0u, 0u, 0u);
      *(uint4*)&k_lds[c][cc] = z;
    }
  }
  // ---- hr (64 x 64): 512 16B chunks, async ----
#pragma unroll
  for (int s = 0; s < 4; ++s) {
    const int lin = tid + s * 128;
    const int c   = lin >> 3;
    const int cc  = (lin & 7) * 8;
    async_cp16(hrbf + (size_t)c * HDC + h * D_HEADC + cc, &hr_lds[c][cc]);
  }
  // ---- V transposed into [dh][c]; bf16 pair loads, scalar scatter ----
#pragma unroll
  for (int s = 0; s < 16; ++s) {
    const int lin  = tid + s * 128;
    const int c    = lin >> 4;          // 0..127
    const int d4   = (lin & 15) * 4;    // 0..60
    const int trow = t0 + c;
    uint2 u = make_uint2(0u, 0u);
    if (trow < T_LENC)
      u = *(const uint2*)(kvbf + ((size_t)b * T_LENC + trow) * (2 * HDC)
                          + HDC + h * D_HEADC + d4);
    v_lds[d4 + 0][c] = (unsigned short)(u.x);
    v_lds[d4 + 1][c] = (unsigned short)(u.x >> 16);
    v_lds[d4 + 2][c] = (unsigned short)(u.y);
    v_lds[d4 + 3][c] = (unsigned short)(u.y >> 16);
  }
  // zero V columns 128..143 (read by wave 3's zero-weight K-chunk)
#pragma unroll
  for (int s = 0; s < 8; ++s) {
    const int lin = tid + s * 128;
    v_lds[lin >> 4][128 + (lin & 15)] = 0;
  }
  async_wait();
  __syncthreads();

  // ---- hq A fragments: direct bf16 uint4 loads ----
  BF16Frag aq[2];
  {
    const unsigned short* hp =
        hqbf + ((size_t)b * Q_LENC + qbase + 16 * wv + l16) * HDC + h * D_HEADC;
#pragma unroll
    for (int kc = 0; kc < 2; ++kc) {
      aq[kc].q[0] = *(const uint4*)(hp + 32 * kc + 8 * sel);
      aq[kc].q[1] = *(const uint4*)(hp + 32 * kc + 16 + 8 * sel);
    }
  }

  // ---- scores S = Q x K^T : 5 column tiles (dense 16 x 80) ----
#pragma unroll
  for (int nt = 0; nt < 5; ++nt) {
    v8f acc = {};
#pragma unroll
    for (int kc = 0; kc < 2; ++kc) {
      BF16Frag bk;
      const unsigned short* p = &k_lds[16 * wv + 16 * nt + l16][0];
      bk.q[0] = *(const uint4*)(p + 32 * kc + 8 * sel);
      bk.q[1] = *(const uint4*)(p + 32 * kc + 16 + 8 * sel);
      acc = wmma_bf16(aq[kc], bk, acc);
    }
#pragma unroll
    for (int v = 0; v < 8; ++v)
      s_lds[wv][v + 8 * sel][16 * nt + l16] = acc[v];
  }
  __syncthreads();

  // ---- relative term Q x hr^T added into the band ----
#pragma unroll
  for (int nt = 0; nt < 4; ++nt) {
    v8f acc = {};
#pragma unroll
    for (int kc = 0; kc < 2; ++kc) {
      BF16Frag bh;
      const unsigned short* p = &hr_lds[16 * nt + l16][0];
      bh.q[0] = *(const uint4*)(p + 32 * kc + 8 * sel);
      bh.q[1] = *(const uint4*)(p + 32 * kc + 16 + 8 * sel);
      acc = wmma_bf16(aq[kc], bh, acc);
    }
    const int j = 16 * nt + l16;
#pragma unroll
    for (int v = 0; v < 8; ++v) {
      const int m = v + 8 * sel;
      s_lds[wv][m][m + j] += acc[v];
    }
  }
  __syncthreads();

  // ---- softmax over the 64-wide band (lanes 0-15: one row each) ----
  const float scale = 0.125f; // 64^-0.5
  float mx = -3.0e38f, ssum = 0.f;
  if (lane >= 16) {
    const int r = lane - 16;
#pragma unroll
    for (int c = 0; c < 104; ++c) wd_lds[wv][r][c] = 0;
  } else {
    const int i  = lane;
    const int qg = qbase + 16 * wv + i;
    for (int j = 0; j < CTXC; ++j) {
      float v = s_lds[wv][i][i + j] * scale;
      const int t = qg + j;
      if (t < CTXC - 1 && maskb[b * (CTXC - 1) + t] != 0.f) v = -3.0e38f;
      s_lds[wv][i][i + j] = v;
      mx = fmaxf(mx, v);
    }
    for (int j = 0; j < CTXC; ++j) {
      float e = __expf(s_lds[wv][i][i + j] - mx);
      s_lds[wv][i][i + j] = e;
      ssum += e;
    }
  }
  __syncthreads();
  if (lane < 16) {
    const int i = lane;
    const float inv = 1.0f / ssum;
    for (int j = 0; j < CTXC; ++j)
      wd_lds[wv][i][i + j] = f2bf(s_lds[wv][i][i + j] * inv);
  }
  __syncthreads();

  // ---- out = Wd(16 x 96 banded) x V(96 x 64) ----
  BF16Frag aw[3];
#pragma unroll
  for (int kc = 0; kc < 3; ++kc) {
    const unsigned short* p = &wd_lds[wv][l16][0];
    aw[kc].q[0] = *(const uint4*)(p + 32 * kc + 8 * sel);
    aw[kc].q[1] = *(const uint4*)(p + 32 * kc + 16 + 8 * sel);
  }
#pragma unroll
  for (int nt = 0; nt < 4; ++nt) {
    v8f acc = {};
#pragma unroll
    for (int kc = 0; kc < 3; ++kc) {
      BF16Frag bv;
      const unsigned short* p = &v_lds[16 * nt + l16][0];
      const int co = 16 * wv + 32 * kc + 8 * sel;
      bv.q[0] = *(const uint4*)(p + co);
      bv.q[1] = *(const uint4*)(p + co + 16);
      acc = wmma_bf16(aw[kc], bv, acc);
    }
#pragma unroll
    for (int v = 0; v < 8; ++v) {
      const int qg = qbase + 16 * wv + v + 8 * sel;
      obf[((size_t)b * Q_LENC + qg) * HDC + h * D_HEADC + 16 * nt + l16]
          = f2bf(acc[v]);
    }
  }
}

// ---------------------------------------------------------------------------
extern "C" void kernel_launch(void* const* d_in, const int* in_sizes, int n_in,
                              void* d_out, int out_size, void* d_ws, size_t ws_size,
                              hipStream_t stream) {
  const float* x   = (const float*)d_in[0];
  const float* r   = (const float*)d_in[1];
  const float* Wkv = (const float*)d_in[2];
  const float* Wq  = (const float*)d_in[3];
  const float* bq  = (const float*)d_in[4];
  const float* Wr  = (const float*)d_in[5];
  const float* Wo  = (const float*)d_in[6];
  float* C_out = (float*)d_out;

  // ---- workspace layout (64-element aligned bf16 buffers) ----
  float* maskbuf = (float*)d_ws;                    // 126 floats (+pad)
  unsigned short* base = (unsigned short*)((char*)d_ws + 1024);
  size_t off = 0;
  auto alloc = [&](size_t elems) {
    unsigned short* p = base + off;
    off += (elems + 63) & ~(size_t)63;
    return p;
  };
  unsigned short* xbf   = alloc((size_t)BATCHC * T_LENC * D_MODELC);
  unsigned short* rbf   = alloc((size_t)CTXC * D_MODELC);
  unsigned short* WkvT  = alloc((size_t)(2 * HDC) * D_MODELC);
  unsigned short* WqT   = alloc((size_t)HDC * D_MODELC);
  unsigned short* WrT   = alloc((size_t)HDC * D_MODELC);
  unsigned short* WoT   = alloc((size_t)D_MODELC * HDC);
  unsigned short* kvbf  = alloc((size_t)BATCHC * T_LENC * (2 * HDC));
  unsigned short* hqbf  = alloc((size_t)BATCHC * Q_LENC * HDC);
  unsigned short* hrbf  = alloc((size_t)CTXC * HDC);
  unsigned short* obf   = alloc((size_t)BATCHC * Q_LENC * HDC);

  // ---- prep: conversions, weight transposes, pad mask ----
  {
    int n4 = (BATCHC * T_LENC * D_MODELC) / 4;
    cvt_bf16_kernel<<<(n4 + 255) / 256, 256, 0, stream>>>(x, xbf, n4);
  }
  {
    int n4 = (CTXC * D_MODELC) / 4;
    cvt_bf16_kernel<<<(n4 + 255) / 256, 256, 0, stream>>>(r, rbf, n4);
  }
  transpose_bf16_kernel<<<dim3((2 * HDC) / 32, D_MODELC / 32), 256, 0, stream>>>(
      Wkv, WkvT, D_MODELC, 2 * HDC);
  transpose_bf16_kernel<<<dim3(HDC / 32, D_MODELC / 32), 256, 0, stream>>>(
      Wq, WqT, D_MODELC, HDC);
  transpose_bf16_kernel<<<dim3(HDC / 32, D_MODELC / 32), 256, 0, stream>>>(
      Wr, WrT, D_MODELC, HDC);
  transpose_bf16_kernel<<<dim3(D_MODELC / 32, HDC / 32), 256, 0, stream>>>(
      Wo, WoT, HDC, D_MODELC);
  mask_kernel<<<1, 128, 0, stream>>>(x, maskbuf);

  // kv = x @ Wkv -> bf16 mirror only
  {
    dim3 g((2 * HDC) / TBN, (BATCHC * T_LENC + TBM - 1) / TBM);
    gemm_bf16<<<g, 256, 0, stream>>>(xbf, WkvT, nullptr, nullptr, kvbf,
        BATCHC * T_LENC, 2 * HDC, D_MODELC,
        T_LENC, (long long)T_LENC * D_MODELC, 0);
  }
  // hq = x[:, 63:] @ Wq + bq -> bf16 mirror only
  {
    dim3 g(HDC / TBN, (BATCHC * Q_LENC) / TBM);
    gemm_bf16<<<g, 256, 0, stream>>>(xbf, WqT, bq, nullptr, hqbf,
        BATCHC * Q_LENC, HDC, D_MODELC,
        Q_LENC, (long long)T_LENC * D_MODELC, CTXC - 1);
  }
  // hr = r @ Wr -> bf16 mirror only
  {
    dim3 g(HDC / TBN, 1);
    gemm_bf16<<<g, 256, 0, stream>>>(rbf, WrT, nullptr, nullptr, hrbf,
        CTXC, HDC, D_MODELC, CTXC, 0, 0);
  }
  // attention -> bf16 obf
  attn_kernel<<<BATCHC * N_HEADC * (Q_LENC / 64), 128, 0, stream>>>(
      kvbf, hqbf, hrbf, maskbuf, obf);
  // final projection: obf @ Wo -> fp32 d_out
  {
    dim3 g(D_MODELC / TBN, (BATCHC * Q_LENC) / TBM);
    gemm_bf16<<<g, 256, 0, stream>>>(obf, WoT, nullptr, C_out, nullptr,
        BATCHC * Q_LENC, D_MODELC, HDC, BATCHC * Q_LENC, 0, 0);
  }
}
